// ISTFA_77429670412761
// MI455X (gfx1250) — matrix operand order repaired and checked by reading, not compiled, not run on hardware
//
#include <hip/hip_runtime.h>

// ---------------- types ----------------
typedef __attribute__((ext_vector_type(16))) __bf16       bf16x16;
typedef __attribute__((ext_vector_type(8)))  float        f32x8;
typedef __attribute__((ext_vector_type(4)))  float        f32x4;
typedef __attribute__((ext_vector_type(4)))  unsigned int u32x4;
typedef __attribute__((ext_vector_type(2)))  unsigned int u32x2;
typedef int v4i_vs __attribute__((vector_size(16)));   // matches builtin param type

#define N_PTS  8192
#define DIM    512
#define NBATCH 16
#define EPSF   1e-8f

#if defined(__gfx1250__) && __has_builtin(__builtin_amdgcn_global_load_async_to_lds_b128)
#define USE_ASYNC_LDS 1
#else
#define USE_ASYNC_LDS 0
#endif

union AFrag { u32x4 q[2]; bf16x16 v; };

// ---------------- helpers ----------------
__device__ __forceinline__ unsigned short f2bf(float f) {
  unsigned int u = __float_as_uint(f);
  u += 0x7FFFu + ((u >> 16) & 1u);          // round-to-nearest-even
  return (unsigned short)(u >> 16);
}
__device__ __forceinline__ float bf2f(unsigned short h) {
  return __uint_as_float(((unsigned int)h) << 16);
}
__device__ __forceinline__ float wsum(float v) {
  for (int o = 16; o; o >>= 1) v += __shfl_xor(v, o, 32);
  return v;
}
__device__ __forceinline__ float wminf(float v) {
  for (int o = 16; o; o >>= 1) v = fminf(v, __shfl_xor(v, o, 32));
  return v;
}
__device__ __forceinline__ int wmini(int v) {
  for (int o = 16; o; o >>= 1) { int t = __shfl_xor(v, o, 32); v = t < v ? t : v; }
  return v;
}

// 16-byte async DMA global -> LDS (ASYNCcnt) with synchronous fallback
__device__ __forceinline__ void cp16(const unsigned short* g, unsigned short* l) {
#if USE_ASYNC_LDS
  __builtin_amdgcn_global_load_async_to_lds_b128(
      (__attribute__((address_space(1))) v4i_vs*)(unsigned long long)g,
      (__attribute__((address_space(3))) v4i_vs*)(unsigned)(unsigned long long)l,
      0, 0);
#else
  *(u32x4*)l = *(const u32x4*)g;
#endif
}
__device__ __forceinline__ void wait_async6() {
#if USE_ASYNC_LDS
#if __has_builtin(__builtin_amdgcn_s_wait_asynccnt)
  __builtin_amdgcn_s_wait_asynccnt(6);
#else
  asm volatile("s_wait_asynccnt 0x6" ::: "memory");
#endif
#endif
}
__device__ __forceinline__ void wait_async0() {
#if USE_ASYNC_LDS
#if __has_builtin(__builtin_amdgcn_s_wait_asynccnt)
  __builtin_amdgcn_s_wait_asynccnt(0);
#else
  asm volatile("s_wait_asynccnt 0x0" ::: "memory");
#endif
#endif
}

// ---------------- K0: zero accumulators ----------------
__global__ void k_init(float* __restrict__ musum, float* __restrict__ scal) {
  int t = threadIdx.x;
  if (t < DIM) musum[t] = 0.f;
  if (t < 16)  scal[t]  = 0.f;
}

// ---------------- K1: z = mean_b(x); split into bf16 hi/lo ----------------
__global__ __launch_bounds__(256) void k_mean(const float* __restrict__ x,
                                              float* __restrict__ z,
                                              unsigned short* __restrict__ zh,
                                              unsigned short* __restrict__ zl) {
  const size_t NDV4 = (size_t)N_PTS * DIM / 4;
  size_t vid = (size_t)blockIdx.x * 256 + threadIdx.x;    // < 1,048,576
  const f32x4* xv = (const f32x4*)x;
  f32x4 s = {0.f, 0.f, 0.f, 0.f};
#pragma unroll
  for (int b = 0; b < NBATCH; b++) s += xv[(size_t)b * NDV4 + vid];
  s *= (1.0f / NBATCH);
  ((f32x4*)z)[vid] = s;
  unsigned short h[4], l[4];
#pragma unroll
  for (int c = 0; c < 4; c++) {
    unsigned short hs = f2bf(s[c]);
    h[c] = hs;
    l[c] = f2bf(s[c] - bf2f(hs));
  }
  u32x2 hp = { (unsigned)h[0] | ((unsigned)h[1] << 16), (unsigned)h[2] | ((unsigned)h[3] << 16) };
  u32x2 lp = { (unsigned)l[0] | ((unsigned)l[1] << 16), (unsigned)l[2] | ((unsigned)l[3] << 16) };
  *(u32x2*)(zh + vid * 4) = hp;
  *(u32x2*)(zl + vid * 4) = lp;
}

// ---------------- K2: column sums of z (for mu) ----------------
__global__ __launch_bounds__(256) void k_colsum(const float* __restrict__ z,
                                                float* __restrict__ musum) {
  int tx = threadIdx.x & 63, ty = threadIdx.x >> 6;      // 64 cols x 4 row-partitions
  int d  = blockIdx.x * 64 + tx;
  int r0 = blockIdx.y * 1024;
  float s = 0.f;
  for (int r = ty; r < 1024; r += 4) s += z[(size_t)(r0 + r) * DIM + d];
  __shared__ float red[256];
  red[threadIdx.x] = s;
  __syncthreads();
  if (ty == 0) atomicAdd(&musum[d], red[tx] + red[tx + 64] + red[tx + 128] + red[tx + 192]);
}

// ---------------- K3: finalize mu, m = mu.mu ----------------
__global__ void k_finmu(float* __restrict__ mu, float* __restrict__ scal) {
  int d = threadIdx.x;                    // 512 threads
  float m = mu[d] * (1.0f / N_PTS);
  mu[d] = m;
  __shared__ float red[512];
  red[d] = m * m;
  __syncthreads();
  for (int o = 256; o; o >>= 1) { if (d < o) red[d] += red[d + o]; __syncthreads(); }
  if (d == 0) scal[0] = red[0];
}

// ---------------- K4: per-row sq[i]=||z_i||^2, v[i]=z_i.mu ----------------
__global__ __launch_bounds__(256) void k_rowstats(const float* __restrict__ z,
                                                  const float* __restrict__ mu,
                                                  float* __restrict__ sq,
                                                  float* __restrict__ v) {
  int i    = blockIdx.x * 8 + (threadIdx.x >> 5);
  int lane = threadIdx.x & 31;
  const f32x4* zr = (const f32x4*)(z + (size_t)i * DIM);
  const f32x4* m4 = (const f32x4*)mu;
  float s = 0.f, w = 0.f;
  for (int c = lane; c < DIM / 4; c += 32) {
    f32x4 a = zr[c], u = m4[c];
    s += a[0]*a[0] + a[1]*a[1] + a[2]*a[2] + a[3]*a[3];
    w += a[0]*u[0] + a[1]*u[1] + a[2]*u[2] + a[3]*u[3];
  }
  s = wsum(s); w = wsum(w);
  if (lane == 0) { sq[i] = s; v[i] = w; }
}

// ---------------- K5: G = z.z^T via split-bf16 WMMA ----------------
// Block tile 128(M) x 64(N), 8 waves (4x2), wave tile 32x32 = 2x2 WMMA tiles.
// G ~= hi.hi + hi.lo + lo.hi  (drop lo.lo, ~2^-16 rel error).
// Double-buffered LDS; staging via GLOBAL_LOAD_ASYNC_TO_LDS_B128 (branch-free:
// 6 straight-line 16B DMAs per thread per K-step).
// LDS buffer layout (ushort units): Ah @ 0 (128x32), Al @ 4096, Bh @ 8192 (64x32),
// Bl @ 10240; buffer stride 12288 (24 KB), total 48 KB.
__global__ __launch_bounds__(256) void k_gram(const unsigned short* __restrict__ zh,
                                              const unsigned short* __restrict__ zl,
                                              float* __restrict__ G) {
  __shared__ __align__(16) unsigned short smem[2 * 12288];
  const int tid  = threadIdx.x;
  const int lane = tid & 31;
  const int w    = tid >> 5;
  const int wm   = w & 3;           // 4 wave-rows along M
  const int wn   = w >> 2;          // 2 wave-cols along N
  const int m0   = blockIdx.x * 128;
  const int n0   = blockIdx.y * 64;
  const int lh   = lane & 15;
  const int hiid = lane >> 4;       // half-wave id
  const int k0a  = hiid * 8;        // A: lane<16 -> K{0..7,16..23}; else K{8..15,24..31}
  const int k0b  = hiid * 8;        // B: lane<16 -> K{0..7,16..23}; else K{8..15,24..31}

  // static per-thread copy assignments: row rr, 16B segment sg (ushort units)
  const int rr  = tid >> 2;                 // 0..63
  const int sgu = (tid & 3) * 8;            // 0,8,16,24 ushorts = 0..48 bytes
  const unsigned short* gAh0 = zh + (size_t)(m0 + rr)      * DIM + sgu;
  const unsigned short* gAh1 = zh + (size_t)(m0 + 64 + rr) * DIM + sgu;
  const unsigned short* gAl0 = zl + (size_t)(m0 + rr)      * DIM + sgu;
  const unsigned short* gAl1 = zl + (size_t)(m0 + 64 + rr) * DIM + sgu;
  const unsigned short* gBh  = zh + (size_t)(n0 + rr)      * DIM + sgu;
  const unsigned short* gBl  = zl + (size_t)(n0 + rr)      * DIM + sgu;
  const int dAh0 = rr * 32 + sgu;
  const int dAh1 = (64 + rr) * 32 + sgu;
  const int dAl0 = 4096 + dAh0;
  const int dAl1 = 4096 + dAh1;
  const int dBh  = 8192 + dAh0;
  const int dBl  = 10240 + dAh0;

  auto issue = [&](int s) {
    const int kb   = s * 32;
    const int base = (s & 1) * 12288;
    cp16(gAh0 + kb, &smem[base + dAh0]);
    cp16(gAh1 + kb, &smem[base + dAh1]);
    cp16(gAl0 + kb, &smem[base + dAl0]);
    cp16(gAl1 + kb, &smem[base + dAl1]);
    cp16(gBh  + kb, &smem[base + dBh ]);
    cp16(gBl  + kb, &smem[base + dBl ]);
  };

  f32x8 acc[2][2];
#pragma unroll
  for (int a = 0; a < 2; a++)
#pragma unroll
    for (int b = 0; b < 2; b++) {
      f32x8 zv = {0.f, 0.f, 0.f, 0.f, 0.f, 0.f, 0.f, 0.f};
      acc[a][b] = zv;
    }

  const int nsteps = DIM / 32;        // 16
  issue(0);
  for (int s = 0; s < nsteps; s++) {
    if (s + 1 < nsteps) { issue(s + 1); wait_async6(); }
    else                { wait_async0(); }
    __syncthreads();                  // current buffer visible to all waves

    const unsigned short* buf = &smem[(s & 1) * 12288];
    AFrag ah[2], al[2], bh[2], bl[2];
#pragma unroll
    for (int t = 0; t < 2; t++) {
      int ra = (wm * 32 + t * 16 + lh) * 32;
      ah[t].q[0] = *(const u32x4*)&buf[ra + k0a];
      ah[t].q[1] = *(const u32x4*)&buf[ra + k0a + 16];
      al[t].q[0] = *(const u32x4*)&buf[4096 + ra + k0a];
      al[t].q[1] = *(const u32x4*)&buf[4096 + ra + k0a + 16];
      int rb = (wn * 32 + t * 16 + lh) * 32;
      bh[t].q[0] = *(const u32x4*)&buf[8192 + rb + k0b];
      bh[t].q[1] = *(const u32x4*)&buf[8192 + rb + k0b + 16];
      bl[t].q[0] = *(const u32x4*)&buf[10240 + rb + k0b];
      bl[t].q[1] = *(const u32x4*)&buf[10240 + rb + k0b + 16];
    }
#pragma unroll
    for (int a = 0; a < 2; a++)
#pragma unroll
      for (int b = 0; b < 2; b++) {
        acc[a][b] = __builtin_amdgcn_wmma_f32_16x16x32_bf16(false, ah[a].v, false, bh[b].v,
                                                            (short)0, acc[a][b], false, false);
        acc[a][b] = __builtin_amdgcn_wmma_f32_16x16x32_bf16(false, ah[a].v, false, bl[b].v,
                                                            (short)0, acc[a][b], false, false);
        acc[a][b] = __builtin_amdgcn_wmma_f32_16x16x32_bf16(false, al[a].v, false, bh[b].v,
                                                            (short)0, acc[a][b], false, false);
      }
    __syncthreads();                  // all reads done before buffer is overwritten
  }
  // C/D layout: lane(0..15)->N, VGPR r -> M=r (lanes<16) / M=8+r (lanes>=16)
#pragma unroll
  for (int a = 0; a < 2; a++)
#pragma unroll
    for (int b = 0; b < 2; b++) {
      int mb = m0 + wm * 32 + a * 16 + hiid * 8;
      int nb = n0 + wn * 32 + b * 16 + lh;
#pragma unroll
      for (int r = 0; r < 8; r++)
        G[(size_t)(mb + r) * N_PTS + nb] = acc[a][b][r];
    }
}

// ---------------- K6: pass1 — rowsum(Kc), Frobenius ss, kNN threshold ----------------
__global__ __launch_bounds__(256) void k_pass1(const float* __restrict__ G,
                                               const float* __restrict__ sq,
                                               const float* __restrict__ v,
                                               float* __restrict__ scal,
                                               float* __restrict__ rsk,
                                               float* __restrict__ tthr) {
  int i    = blockIdx.x * 8 + (threadIdx.x >> 5);
  int lane = threadIdx.x & 31;
  float m  = scal[0];
  float vi = v[i], si = sq[i];
  const f32x4* Gr = (const f32x4*)(G + (size_t)i * N_PTS);
  const f32x4* v4 = (const f32x4*)v;
  const f32x4* q4 = (const f32x4*)sq;
  float top[8];
#pragma unroll
  for (int r = 0; r < 8; r++) top[r] = __builtin_inff();
  float s = 0.f, ss = 0.f;
  for (int j = lane; j < N_PTS / 4; j += 32) {
    if (j + 256 < N_PTS / 4) __builtin_prefetch((const void*)(Gr + j + 256), 0, 0);
    f32x4 g = Gr[j], vj = v4[j], qj = q4[j];
#pragma unroll
    for (int c = 0; c < 4; c++) {
      float kc = g[c] - vi - vj[c] + m;          // centered Gram element
      s += kc; ss += kc * kc;
      float d2 = fmaxf(si + qj[c] - 2.0f * g[c], 0.0f);
      if (d2 < top[7]) {                          // sorted ascending insert
        top[7] = d2;
#pragma unroll
        for (int k = 7; k > 0; k--) {
          float a = top[k - 1], b = top[k];
          top[k - 1] = fminf(a, b); top[k] = fmaxf(a, b);
        }
      }
    }
  }
  s = wsum(s); ss = wsum(ss);
  // merge 32 sorted 8-lists: 8 rounds of wave-min + single-leader pop
  float kth = 0.f;
#pragma unroll
  for (int r = 0; r < 8; r++) {
    float gmin = wminf(top[0]);
    kth = gmin;
    int cand = (top[0] == gmin) ? lane : 64;
    int lead = wmini(cand);
    if (lane == lead) {
#pragma unroll
      for (int k = 0; k < 7; k++) top[k] = top[k + 1];
      top[7] = __builtin_inff();
    }
  }
  if (lane == 0) { rsk[i] = s; tthr[i] = kth; atomicAdd(&scal[1], ss); }
}

// ---------------- K7: derived scalars ----------------
__global__ void k_scal(const float* __restrict__ rsk, float* __restrict__ scal) {
  int t = threadIdx.x;
  float s = 0.f;
  for (int i = t; i < N_PTS; i += 256) s += rsk[i];
  __shared__ float red[256];
  red[t] = s; __syncthreads();
  for (int o = 128; o; o >>= 1) { if (t < o) red[t] += red[t + o]; __syncthreads(); }
  if (t == 0) {
    float F     = sqrtf(scal[1]) + EPSF;         // ||Kc||_F + eps
    float invF  = 1.0f / F;
    scal[2] = invF;
    scal[3] = invF / (float)N_PTS;               // invNF
    scal[4] = red[0] * invF / ((float)N_PTS * (float)N_PTS);  // cbar
  }
}

// ---------------- K8: pass2 — L1, rowsum, knn count per row ----------------
__global__ __launch_bounds__(256) void k_pass2(const float* __restrict__ G,
                                               const float* __restrict__ sq,
                                               const float* __restrict__ v,
                                               const float* __restrict__ rsk,
                                               const float* __restrict__ tthr,
                                               const float* __restrict__ scal,
                                               float* __restrict__ L1,
                                               float* __restrict__ RS2,
                                               float* __restrict__ CNT) {
  int i    = blockIdx.x * 8 + (threadIdx.x >> 5);
  int lane = threadIdx.x & 31;
  float m = scal[0], invF = scal[2], invNF = scal[3], cbar = scal[4];
  float vi = v[i], si = sq[i], ti = tthr[i], ri = rsk[i];
  const f32x4* Gr = (const f32x4*)(G + (size_t)i * N_PTS);
  const f32x4* v4 = (const f32x4*)v;
  const f32x4* q4 = (const f32x4*)sq;
  const f32x4* r4 = (const f32x4*)rsk;
  float l1 = 0.f, r2 = 0.f, cnt = 0.f;
  for (int j = lane; j < N_PTS / 4; j += 32) {
    if (j + 256 < N_PTS / 4) __builtin_prefetch((const void*)(Gr + j + 256), 0, 0);
    f32x4 g = Gr[j], vj = v4[j], qj = q4[j], rj = r4[j];
#pragma unroll
    for (int c = 0; c < 4; c++) {
      float kc  = g[c] - vi - vj[c] + m;
      float kpp = kc * invF - (ri + rj[c]) * invNF + cbar;   // double-centered K
      l1 += fabsf(kpp); r2 += kpp;
      float d2 = fmaxf(si + qj[c] - 2.0f * g[c], 0.0f);
      cnt += (d2 <= ti) ? 1.f : 0.f;
    }
  }
  l1 = wsum(l1); r2 = wsum(r2); cnt = wsum(cnt);
  if (lane == 0) { L1[i] = l1; RS2[i] = r2; CNT[i] = cnt; }
}

// ---------------- K9: pass3 — row L1 of mixed A ----------------
__global__ __launch_bounds__(256) void k_pass3(const float* __restrict__ G,
                                               const float* __restrict__ sq,
                                               const float* __restrict__ v,
                                               const float* __restrict__ rsk,
                                               const float* __restrict__ tthr,
                                               const float* __restrict__ scal,
                                               const float* __restrict__ L1,
                                               const float* __restrict__ RS2,
                                               const float* __restrict__ CNT,
                                               float* __restrict__ invA) {
  int i    = blockIdx.x * 8 + (threadIdx.x >> 5);
  int lane = threadIdx.x & 31;
  float m = scal[0], invF = scal[2], invNF = scal[3], cbar = scal[4];
  float vi = v[i], si = sq[i], ti = tthr[i], ri = rsk[i];
  float il1  = 1.0f / (L1[i] + EPSF);
  float dinv = 1.0f / (RS2[i] * il1 + 0.01f + EPSF);
  float i1   = 1.0f / (CNT[i] + EPSF);
  const f32x4* Gr = (const f32x4*)(G + (size_t)i * N_PTS);
  const f32x4* v4 = (const f32x4*)v;
  const f32x4* q4 = (const f32x4*)sq;
  const f32x4* r4 = (const f32x4*)rsk;
  float acc = 0.f;
  for (int j4 = lane; j4 < N_PTS / 4; j4 += 32) {
    if (j4 + 256 < N_PTS / 4) __builtin_prefetch((const void*)(Gr + j4 + 256), 0, 0);
    f32x4 g = Gr[j4], vj = v4[j4], qj = q4[j4], rj = r4[j4];
#pragma unroll
    for (int c = 0; c < 4; c++) {
      int j = j4 * 4 + c;
      float kc  = g[c] - vi - vj[c] + m;
      float kpp = kc * invF - (ri + rj[c]) * invNF + cbar;
      float a2  = kpp * il1;
      if (j == i) a2 += 0.01f;
      a2 *= dinv;
      float d2 = fmaxf(si + qj[c] - 2.0f * g[c], 0.0f);
      float a1 = (d2 <= ti) ? i1 : 0.f;
      acc += fabsf(0.5f * (a1 + a2));
    }
  }
  acc = wsum(acc);
  if (lane == 0) invA[i] = 1.0f / fmaxf(acc, 1e-12f);
}

// ---------------- K10: pass4 — write final A in place over G ----------------
__global__ __launch_bounds__(256) void k_pass4(float* __restrict__ G,
                                               const float* __restrict__ sq,
                                               const float* __restrict__ v,
                                               const float* __restrict__ rsk,
                                               const float* __restrict__ tthr,
                                               const float* __restrict__ scal,
                                               const float* __restrict__ L1,
                                               const float* __restrict__ RS2,
                                               const float* __restrict__ CNT,
                                               const float* __restrict__ invA) {
  int i    = blockIdx.x * 8 + (threadIdx.x >> 5);
  int lane = threadIdx.x & 31;
  float m = scal[0], invF = scal[2], invNF = scal[3], cbar = scal[4];
  float vi = v[i], si = sq[i], ti = tthr[i], ri = rsk[i];
  float il1  = 1.0f / (L1[i] + EPSF);
  float dinv = 1.0f / (RS2[i] * il1 + 0.01f + EPSF);
  float i1   = 1.0f / (CNT[i] + EPSF);
  float ia   = invA[i];
  f32x4* Gr = (f32x4*)(G + (size_t)i * N_PTS);
  const f32x4* v4 = (const f32x4*)v;
  const f32x4* q4 = (const f32x4*)sq;
  const f32x4* r4 = (const f32x4*)rsk;
  for (int j4 = lane; j4 < N_PTS / 4; j4 += 32) {
    if (j4 + 256 < N_PTS / 4) __builtin_prefetch((const void*)(Gr + j4 + 256), 0, 0);
    f32x4 g = Gr[j4], vj = v4[j4], qj = q4[j4], rj = r4[j4];
    f32x4 o;
#pragma unroll
    for (int c = 0; c < 4; c++) {
      int j = j4 * 4 + c;
      float kc  = g[c] - vi - vj[c] + m;
      float kpp = kc * invF - (ri + rj[c]) * invNF + cbar;
      float a2  = kpp * il1;
      if (j == i) a2 += 0.01f;
      a2 *= dinv;
      float d2 = fmaxf(si + qj[c] - 2.0f * g[c], 0.0f);
      float a1 = (d2 <= ti) ? i1 : 0.f;
      o[c] = 0.5f * (a1 + a2) * ia;
    }
    Gr[j4] = o;   // each element read+written exactly once by this thread
  }
}

// ---------------- host ----------------
extern "C" void kernel_launch(void* const* d_in, const int* in_sizes, int n_in,
                              void* d_out, int out_size, void* d_ws, size_t ws_size,
                              hipStream_t stream) {
  (void)in_sizes; (void)n_in; (void)out_size; (void)ws_size;
  const float* x = (const float*)d_in[0];
  float* G = (float*)d_out;                       // 8192x8192, holds Gram then final A
  char* ws = (char*)d_ws;

  const size_t ND = (size_t)N_PTS * DIM;
  float*          z    = (float*)(ws);                        // 16 MB
  unsigned short* zh   = (unsigned short*)(ws + ND * 4);      //  8 MB
  unsigned short* zl   = (unsigned short*)(ws + ND * 6);      //  8 MB
  size_t base = ND * 8;
  float* mu   = (float*)(ws + base);               base += 4096;
  float* sq   = (float*)(ws + base);               base += (size_t)N_PTS * 4;
  float* v    = (float*)(ws + base);               base += (size_t)N_PTS * 4;
  float* rsk  = (float*)(ws + base);               base += (size_t)N_PTS * 4;
  float* tthr = (float*)(ws + base);               base += (size_t)N_PTS * 4;
  float* L1   = (float*)(ws + base);               base += (size_t)N_PTS * 4;
  float* RS2  = (float*)(ws + base);               base += (size_t)N_PTS * 4;
  float* CNT  = (float*)(ws + base);               base += (size_t)N_PTS * 4;
  float* invA = (float*)(ws + base);               base += (size_t)N_PTS * 4;
  float* scal = (float*)(ws + base);

  k_init   <<<1, 512, 0, stream>>>(mu, scal);
  k_mean   <<<4096, 256, 0, stream>>>(x, z, zh, zl);
  k_colsum <<<dim3(8, 8), 256, 0, stream>>>(z, mu);
  k_finmu  <<<1, 512, 0, stream>>>(mu, scal);
  k_rowstats<<<N_PTS / 8, 256, 0, stream>>>(z, mu, sq, v);
  k_gram   <<<dim3(N_PTS / 128, N_PTS / 64), 256, 0, stream>>>(zh, zl, G);
  k_pass1  <<<N_PTS / 8, 256, 0, stream>>>(G, sq, v, scal, rsk, tthr);
  k_scal   <<<1, 256, 0, stream>>>(rsk, scal);
  k_pass2  <<<N_PTS / 8, 256, 0, stream>>>(G, sq, v, rsk, tthr, scal, L1, RS2, CNT);
  k_pass3  <<<N_PTS / 8, 256, 0, stream>>>(G, sq, v, rsk, tthr, scal, L1, RS2, CNT, invA);
  k_pass4  <<<N_PTS / 8, 256, 0, stream>>>(G, sq, v, rsk, tthr, scal, L1, RS2, CNT, invA);
}
